// QuanvolutionHybrid_65481071405780
// MI455X (gfx1250) — matrix operand order repaired and checked by reading, not compiled, not run on hardware
//
#include <hip/hip_runtime.h>
#include <hip/hip_bf16.h>
#include <math.h>

typedef __attribute__((ext_vector_type(16))) _Float16 v16h;
typedef __attribute__((ext_vector_type(8)))  _Float16 v8h;
typedef __attribute__((ext_vector_type(8)))  float    v8f;

#define B_IMG   8192
#define IMG     28
#define NPIX    784      // 28*28, also recon row length
#define NPATCH  196
#define KDIM    784      // NUM_PATCHES * PATCH_OUT
#define KPAD    800      // KDIM padded to a multiple of 32 for WMMA K-loop
#define NCLS    10

// ---------------- workspace layout (bytes) ----------------
// 0        : 256 f32   per-block partial maxima
// 1024     : 1   f32   final global max
// 2048     : 256 f32   U (16x16 circuit matrix, row-major)
// 4096     : Wr_f16 [784][KPAD]      (1,254,400 B)
// 1258496  : flat_f16 [8192][KPAD]   (13,107,200 B)

// ---------- global max, pass 1: 256 blocks -> 256 partials ----------
__global__ void k_max1(const float* __restrict__ x, float* __restrict__ part, int n) {
    __shared__ float red[256];
    float m = -INFINITY;
    for (int i = blockIdx.x * blockDim.x + threadIdx.x; i < n; i += gridDim.x * blockDim.x)
        m = fmaxf(m, x[i]);
    red[threadIdx.x] = m;
    __syncthreads();
    for (int s = 128; s > 0; s >>= 1) {
        if (threadIdx.x < s) red[threadIdx.x] = fmaxf(red[threadIdx.x], red[threadIdx.x + s]);
        __syncthreads();
    }
    if (threadIdx.x == 0) part[blockIdx.x] = red[0];
}

// ---------- global max, pass 2: 256 partials -> 1 ----------
__global__ void k_max2(const float* __restrict__ part, float* __restrict__ outm) {
    __shared__ float red[256];
    red[threadIdx.x] = part[threadIdx.x];
    __syncthreads();
    for (int s = 128; s > 0; s >>= 1) {
        if (threadIdx.x < s) red[threadIdx.x] = fmaxf(red[threadIdx.x], red[threadIdx.x + s]);
        __syncthreads();
    }
    if (threadIdx.x == 0) outm[0] = red[0];
}

// ---------- build the fixed 16x16 ansatz unitary U ----------
// Thread t simulates the 2-layer circuit applied to basis state |t>, giving column t.
// Wire w <-> bit (3-w) of the 4-bit index (wire 0 is the most significant axis).
__global__ void k_buildU(const float* __restrict__ params, float* __restrict__ U) {
    int t = threadIdx.x;
    if (t >= 16) return;
    float a[16];
    #pragma unroll
    for (int k = 0; k < 16; ++k) a[k] = (k == t) ? 1.0f : 0.0f;
    for (int l = 0; l < 2; ++l) {
        for (int w = 0; w < 4; ++w) {
            float h = 0.5f * params[l * 4 + w];
            float c = cosf(h), s = sinf(h);
            int st = 8 >> w;
            #pragma unroll
            for (int k = 0; k < 16; ++k) {
                if (k & st) continue;
                float a0 = a[k], a1 = a[k + st];
                a[k]      = c * a0 - s * a1;
                a[k + st] = s * a0 + c * a1;
            }
        }
        const int cw[4] = {0, 1, 2, 3}, tw[4] = {1, 2, 3, 0};
        for (int e = 0; e < 4; ++e) {
            int cm = 8 >> cw[e], tm = 8 >> tw[e];
            #pragma unroll
            for (int k = 0; k < 16; ++k) {
                if ((k & cm) && !(k & tm)) {
                    float tmp = a[k]; a[k] = a[k | tm]; a[k | tm] = tmp;
                }
            }
        }
    }
    for (int r = 0; r < 16; ++r) U[r * 16 + t] = a[r];
}

// ---------- convert Wr (f32, 784x784) to f16 with K padded to 800 ----------
__global__ void k_cvt_wr(const float* __restrict__ Wr, _Float16* __restrict__ WrH) {
    int i = blockIdx.x * blockDim.x + threadIdx.x;
    if (i >= KDIM * KPAD) return;
    int n = i / KPAD, k = i % KPAD;
    WrH[i] = (k < KDIM) ? (_Float16)Wr[n * KDIM + k] : (_Float16)0.0f;
}

// ---------- per-image quantum-patch simulation -> flat_f16 [b][KPAD] ----------
__global__ void __launch_bounds__(224)
k_sim(const float* __restrict__ x, const float* __restrict__ U,
      const float* __restrict__ maxv, _Float16* __restrict__ flat) {
    __shared__ float Us[256];
    int tid = threadIdx.x;
    int b   = blockIdx.x;
    for (int i = tid; i < 256; i += 224) Us[i] = U[i];
    __syncthreads();
    if (tid < NPATCH) {
        float inv = 1.0f / maxv[0];
        int py = tid / 14, px = tid % 14;
        const float* xb = x + (size_t)b * NPIX;
        float t0 = xb[(2 * py)     * IMG + 2 * px    ] * inv;
        float t1 = xb[(2 * py)     * IMG + 2 * px + 1] * inv;
        float t2 = xb[(2 * py + 1) * IMG + 2 * px    ] * inv;
        float t3 = xb[(2 * py + 1) * IMG + 2 * px + 1] * inv;
        float s0, c0, s1, c1, s2, c2, s3, c3;
        sincosf(0.5f * t0, &s0, &c0);
        sincosf(0.5f * t1, &s1, &c1);
        sincosf(0.5f * t2, &s2, &c2);
        sincosf(0.5f * t3, &s3, &c3);
        float w0[2] = {c0, s0}, w1[2] = {c1, s1}, w2[2] = {c2, s2}, w3[2] = {c3, s3};
        float psi[16];
        #pragma unroll
        for (int k = 0; k < 16; ++k)
            psi[k] = w0[(k >> 3) & 1] * w1[(k >> 2) & 1] * w2[(k >> 1) & 1] * w3[k & 1];
        float f0 = 0.f, f1 = 0.f, f2 = 0.f, f3 = 0.f;
        #pragma unroll
        for (int r = 0; r < 16; ++r) {
            float v = 0.f;
            #pragma unroll
            for (int c = 0; c < 16; ++c) v += Us[r * 16 + c] * psi[c];
            float p = v * v;
            f0 += ((r >> 3) & 1) ? -p : p;
            f1 += ((r >> 2) & 1) ? -p : p;
            f2 += ((r >> 1) & 1) ? -p : p;
            f3 += (r & 1)        ? -p : p;
        }
        size_t off = (size_t)b * KPAD + tid * 4;
        flat[off + 0] = (_Float16)f0;
        flat[off + 1] = (_Float16)f1;
        flat[off + 2] = (_Float16)f2;
        flat[off + 3] = (_Float16)f3;
    } else if (tid < NPATCH + 16) {
        // zero the K padding columns (ws is poisoned, must rewrite each call)
        flat[(size_t)b * KPAD + KDIM + (tid - NPATCH)] = (_Float16)0.0f;
    }
}

// ---------- classifier: logits + log_softmax, one wave (32 lanes) per row ----------
__global__ void __launch_bounds__(256)
k_cls(const _Float16* __restrict__ flat, const float* __restrict__ Wc,
      const float* __restrict__ bc, float* __restrict__ out) {
    int lane = threadIdx.x & 31;
    int wave = threadIdx.x >> 5;
    int row  = blockIdx.x * 8 + wave;
    const _Float16* fr = flat + (size_t)row * KPAD;
    float acc[NCLS];
    #pragma unroll
    for (int c = 0; c < NCLS; ++c) acc[c] = 0.0f;
    for (int k = lane; k < KDIM; k += 32) {
        float fv = (float)fr[k];
        #pragma unroll
        for (int c = 0; c < NCLS; ++c) acc[c] += fv * Wc[c * KDIM + k];
    }
    #pragma unroll
    for (int c = 0; c < NCLS; ++c)
        for (int off = 16; off > 0; off >>= 1)
            acc[c] += __shfl_xor(acc[c], off, 32);
    if (lane == 0) {
        float l[NCLS], m = -INFINITY;
        #pragma unroll
        for (int c = 0; c < NCLS; ++c) { l[c] = acc[c] + bc[c]; m = fmaxf(m, l[c]); }
        float s = 0.0f;
        #pragma unroll
        for (int c = 0; c < NCLS; ++c) s += expf(l[c] - m);
        float lse = m + logf(s);
        #pragma unroll
        for (int c = 0; c < NCLS; ++c) out[(size_t)row * NCLS + c] = l[c] - lse;
    }
}

// ---------- recon GEMM: out[b][n] = sum_k flat[b][k] * Wr[n][k] + br[n] ----------
// One 16x16 output tile per wave, K loop over KPAD in steps of 32 via
// v_wmma_f32_16x16x32_f16.  Fragment layouts (wave32, CDNA5 ISA 7.12.2):
//   A 16x32 f16 : lane L holds row M=L&15; halves 0..7 = K kOff..kOff+7,
//                 halves 8..15 = K kOff+16..kOff+23, kOff = (L>>4)*8
//   B 32x16 f16 : lane L holds col N=L&15; halves 0..7 = K kOff..kOff+7,
//                 halves 8..15 = K kOff+16..kOff+23, kOff = (L>>4)*8
// Both map to contiguous, 16B-aligned global loads (Wr row n is contiguous in k).
__global__ void __launch_bounds__(256)
k_recon(const _Float16* __restrict__ flat, const _Float16* __restrict__ WrH,
        const float* __restrict__ br, float* __restrict__ out) {
    const int lane  = threadIdx.x & 31;
    const int wave  = threadIdx.x >> 5;
    const int mbase = blockIdx.y * 128 + wave * 16;
    const int nbase = blockIdx.x * 16;

    const int rA = mbase + (lane & 15);
    const int kA = (lane >> 4) * 8;
    const int nB = nbase + (lane & 15);
    const int kB = (lane >> 4) * 8;

    const _Float16* pa = flat + (size_t)rA * KPAD + kA;
    const _Float16* pb = WrH  + (size_t)nB * KPAD + kB;

    v8f acc = {};
    for (int kc = 0; kc < KPAD; kc += 32) {
        v8h a_lo = *(const v8h*)(pa + kc);
        v8h a_hi = *(const v8h*)(pa + kc + 16);
        v8h b_lo = *(const v8h*)(pb + kc);
        v8h b_hi = *(const v8h*)(pb + kc + 16);
        v16h a, bm;
        #pragma unroll
        for (int i = 0; i < 8; ++i) {
            a[i] = a_lo[i];  a[i + 8] = a_hi[i];
            bm[i] = b_lo[i]; bm[i + 8] = b_hi[i];
        }
        acc = __builtin_amdgcn_wmma_f32_16x16x32_f16(
            /*neg_a=*/false, a, /*neg_b=*/false, bm,
            /*c_mod=*/(short)0, acc, /*reuse_a=*/false, /*reuse_b=*/false);
    }
    // D layout: lane L holds col N=L&15; VGPR r holds row (L>>4)*8 + r
    const int   col  = nbase + (lane & 15);
    const float bias = br[col];
    const int   rowb = mbase + (lane >> 4) * 8;
    #pragma unroll
    for (int r = 0; r < 8; ++r)
        out[(size_t)(rowb + r) * NPIX + col] = acc[r] + bias;
}

extern "C" void kernel_launch(void* const* d_in, const int* in_sizes, int n_in,
                              void* d_out, int out_size, void* d_ws, size_t ws_size,
                              hipStream_t stream) {
    const float* x      = (const float*)d_in[0];  // (8192,1,28,28)
    const float* params = (const float*)d_in[1];  // (2,4)
    const float* Wc     = (const float*)d_in[2];  // (10,784)
    const float* bc     = (const float*)d_in[3];  // (10,)
    const float* Wr     = (const float*)d_in[4];  // (784,784)
    const float* br     = (const float*)d_in[5];  // (784,)

    float* out_logits = (float*)d_out;                       // 8192*10
    float* out_recon  = out_logits + (size_t)B_IMG * NCLS;   // 8192*784

    char*     ws    = (char*)d_ws;
    float*    part  = (float*)ws;
    float*    maxv  = (float*)(ws + 1024);
    float*    U     = (float*)(ws + 2048);
    _Float16* WrH   = (_Float16*)(ws + 4096);
    _Float16* flat  = (_Float16*)(ws + 4096 + (size_t)KDIM * KPAD * sizeof(_Float16));

    k_max1  <<<256, 256, 0, stream>>>(x, part, B_IMG * NPIX);
    k_max2  <<<1, 256, 0, stream>>>(part, maxv);
    k_buildU<<<1, 32, 0, stream>>>(params, U);
    k_cvt_wr<<<(KDIM * KPAD + 255) / 256, 256, 0, stream>>>(Wr, WrH);
    k_sim   <<<B_IMG, 224, 0, stream>>>(x, U, maxv, flat);
    k_cls   <<<B_IMG / 8, 256, 0, stream>>>(flat, Wc, bc, out_logits);
    dim3 g(NPIX / 16, B_IMG / 128);
    k_recon <<<g, 256, 0, stream>>>(flat, WrH, br, out_recon);
}
